// TensorProductWithSphericalHarmonicsLinear_57664230916637
// MI455X (gfx1250) — matrix operand order repaired and checked by reading, not compiled, not run on hardware
//
#include <hip/hip_runtime.h>

// MI455X / gfx1250, wave32.
//   K-A: W fp32 -> bf16 (512KB, stays hot in 192MB L2).
//   K-B: per-sample Wigner D blocks (exact real-basis generators; expm via
//        scaling-and-squaring) -> ws (84 f32 / sample).
//   K-C: fused [x @ Dblk] -> bf16 LDS tile -> v_wmma_f32_16x16x32_bf16 GEMM
//        vs W(bf16, L2) -> LDS -> [y @ Dblk^T] -> f32 out.
// ROWS=32 / block, each wave owns 4 unique N-tiles x both M-tiles so W is
// read exactly once per block: total W L2 traffic = 4096 * 512KB = 2GB,
// against a ~0.63GB HBM floor (~27us @ 23.3TB/s). bf16 WMMA (K=32) keeps the
// 68.7GFLOP GEMM near that floor (fp32 WMMA K=4 would be ~8x more math time).

typedef __attribute__((ext_vector_type(16))) __bf16  v16bf;
typedef __attribute__((ext_vector_type(8)))  float   v8f;

#define DIMF    512
#define ROWS    32          // samples per block in fused kernel
#define LDA     520         // bf16 elems per staged A row (LDS bank padding)
#define LDCH    516         // f32 elems per staged C row (16-row phase buffer)
#define NTHR    256         // 8 wave32

__device__ __forceinline__ unsigned short f2bf(float f) {
  unsigned int u = __float_as_uint(f);                 // round-to-nearest-even
  return (unsigned short)((u + 0x7FFFu + ((u >> 16) & 1u)) >> 16);
}

// ---------------- Kernel A: W fp32 -> bf16 ----------------
__global__ void wcvt_kernel(const float* __restrict__ W,
                            unsigned short* __restrict__ Wb) {
  int i = blockIdx.x * blockDim.x + threadIdx.x;
  Wb[i] = f2bf(W[i]);
}

// ---------------- Kernel B: Wigner D ----------------
// Real-SH-basis generators (derived from the reference's Q-conjugated su2 gens):
//   X1[l-n, l+n] = n,  X1[l+n, l-n] = -n           -> exp(a*X1) closed form
//   X0 tridiagonal antisymmetric, b_n = 0.5*sqrt(l(l+1) - n(n+1)),
//     links (l+n <-> l+n+1) and (l-n <-> l-n-1), sqrt(2)*b_0 at the m=0 link.
// D = exp(alpha*X1) * exp(beta*X0).
template<int L>
__device__ void wigner_block(float alpha, float beta, float* __restrict__ out) {
  const int d = 2 * L + 1;
  float X[d * d], R[d * d], T[d * d];

  const float bs = beta * (1.0f / 32.0f);   // 2^-5 scaling
  for (int i = 0; i < d * d; ++i) X[i] = 0.0f;
  for (int n = 0; n < L; ++n) {
    float b = 0.5f * sqrtf((float)(L * (L + 1) - n * (n + 1)));
    if (n == 0) b *= 1.41421356237f;
    X[(L + n + 1) * d + (L + n)] =  b * bs;
    X[(L + n) * d + (L + n + 1)] = -b * bs;
    if (n >= 1) {
      X[(L - n - 1) * d + (L - n)] =  b * bs;
      X[(L - n) * d + (L - n - 1)] = -b * bs;
    }
  }

  // R = expm(X): 6-term Horner Taylor
  for (int i = 0; i < d * d; ++i) R[i] = 0.0f;
  for (int i = 0; i < d; ++i) R[i * d + i] = 1.0f;
  for (int k = 6; k >= 1; --k) {
    for (int i = 0; i < d; ++i)
      for (int j = 0; j < d; ++j) {
        float s = 0.0f;
        for (int q = 0; q < d; ++q) s += X[i * d + q] * R[q * d + j];
        T[i * d + j] = s;
      }
    const float inv = 1.0f / (float)k;
    for (int i = 0; i < d * d; ++i) R[i] = T[i] * inv;
    for (int i = 0; i < d; ++i) R[i * d + i] += 1.0f;
  }
  // 5 squarings
  for (int sq = 0; sq < 5; ++sq) {
    for (int i = 0; i < d; ++i)
      for (int j = 0; j < d; ++j) {
        float s = 0.0f;
        for (int q = 0; q < d; ++q) s += R[i * d + q] * R[q * d + j];
        T[i * d + j] = s;
      }
    for (int i = 0; i < d * d; ++i) R[i] = T[i];
  }

  // out = exp(alpha*X1) * R   (sparse left-multiply)
  for (int j = 0; j < d; ++j) out[L * d + j] = R[L * d + j];
  for (int n = 1; n <= L; ++n) {
    float c = cosf((float)n * alpha), s = sinf((float)n * alpha);
    for (int j = 0; j < d; ++j) {
      float lo = R[(L - n) * d + j], hi = R[(L + n) * d + j];
      out[(L - n) * d + j] =  c * lo + s * hi;
      out[(L + n) * d + j] = -s * lo + c * hi;
    }
  }
}

__global__ void wigner_kernel(const float* __restrict__ vec,
                              float* __restrict__ Dws, int n) {
  int i = blockIdx.x * blockDim.x + threadIdx.x;
  if (i >= n) return;
  float vx = vec[3 * i + 0], vy = vec[3 * i + 1], vz = vec[3 * i + 2];
  float r2 = vx * vx + vy * vy + vz * vz;
  float inv = r2 > 0.0f ? rsqrtf(r2) : 0.0f;
  float cy = fminf(1.0f, fmaxf(-1.0f, vy * inv));
  float beta  = acosf(cy);
  float alpha = atan2f(vx, vz);          // == atan2(x/r, z/r)
  float* o = Dws + (size_t)i * 84;       // [0..8]=D1, [9..33]=D2, [34..82]=D3
  wigner_block<1>(alpha, beta, o + 0);
  wigner_block<2>(alpha, beta, o + 9);
  wigner_block<3>(alpha, beta, o + 34);
}

// ---------------- Kernel C: fused rotate-in / GEMM / rotate-out ------------
__global__ __launch_bounds__(NTHR, 1)
void fused_kernel(const float* __restrict__ x,
                  const unsigned short* __restrict__ Wb,
                  const float* __restrict__ Dws,
                  float* __restrict__ out, int n) {
  __shared__ float sD[ROWS * 84];                              // 10752 B
  __shared__ __align__(16) unsigned char smem[ROWS * LDA * 2]; // 33280 B
  unsigned short* sA = (unsigned short*)smem;  // bf16 A tile [ROWS][LDA]
  float*          sC = (float*)smem;           // f32 C phase  [16][LDCH] (after barrier)

  const int tid  = threadIdx.x;
  const int row0 = blockIdx.x * ROWS;
  if (row0 >= n) return;

  // stage per-row D blocks
  for (int i = tid; i < ROWS * 84; i += NTHR)
    sD[i] = Dws[(size_t)(row0 + i / 84) * 84 + (i % 84)];
  __syncthreads();

  const int lbase[4] = {0, 32, 128, 288};
  const int doff[4]  = {0, 0, 9, 34};

  // prologue: x_rot = x @ blockdiag(D)  (inverse rotation), bf16 into LDS
  for (int c = tid; c < ROWS * 128; c += NTHR) {
    int r = c >> 7, ch = c & 127;
    int l = ch >> 5, u = ch & 31;
    int dd = 2 * l + 1;
    int base = lbase[l] + u * dd;
    const float* xr = x + (size_t)(row0 + r) * DIMF + base;
    unsigned short* ar = sA + r * LDA + base;
    if (l == 0) {
      ar[0] = f2bf(xr[0]);
    } else {
      const float* Db = sD + r * 84 + doff[l];
      float xin[7];
      for (int j = 0; j < dd; ++j) xin[j] = xr[j];
      for (int i2 = 0; i2 < dd; ++i2) {
        float s = 0.0f;
        for (int j = 0; j < dd; ++j) s += xin[j] * Db[j * dd + i2];
        ar[i2] = f2bf(s);
      }
    }
  }
  __syncthreads();

  // GEMM: C[32 x 512] = A[32 x 512](LDS,bf16) * W[512 x 512](L2,bf16)
  // Each wave: 4 unique N-tiles x 2 M-tiles -> each B fragment feeds 2 WMMAs,
  // so the block reads W exactly once (no cross-wave duplication).
  const int wave  = tid >> 5, lane = tid & 31;
  const int mrow  = lane & 15;
  const int khalf = (lane >> 4) << 3;     // A: lanes 16-31 take K+8
  const v8f vzero = {0, 0, 0, 0, 0, 0, 0, 0};
  v8f acc[2][4];
  for (int mt = 0; mt < 2; ++mt)
    for (int j = 0; j < 4; ++j) acc[mt][j] = vzero;

  for (int k0 = 0; k0 < DIMF; k0 += 32) {
    // A fragments (16x32 bf16 each M-tile), ISA 7.12.2 layout
    union { v16bf v; unsigned int u[8]; } a[2];
    const int kof[8] = {0, 2, 4, 6, 16, 18, 20, 22};
#pragma unroll
    for (int mt = 0; mt < 2; ++mt) {
      const unsigned short* arow = sA + (mt * 16 + mrow) * LDA + k0 + khalf;
#pragma unroll
      for (int v = 0; v < 8; ++v)
        a[mt].u[v] = *(const unsigned int*)(arow + kof[v]);
    }

    const unsigned short* wrow = Wb + (size_t)(k0 + lane) * DIMF;
    if (k0 + 32 < DIMF) __builtin_prefetch(wrow + 32 * DIMF, 0, 1);

#pragma unroll
    for (int j = 0; j < 4; ++j) {
      int n0 = ((wave << 2) + j) << 4;    // N-tile origin (unique per wave)
      union { v16bf v; unsigned int u[8]; } b;  // B: N=lane&15, K-pair=khalf+kof[v],+1
#pragma unroll
      for (int v2 = 0; v2 < 8; ++v2) {
        const unsigned short* bp = Wb + (size_t)(k0 + khalf + kof[v2]) * DIMF + n0 + mrow;
        b.u[v2] = (unsigned int)bp[0] | ((unsigned int)bp[DIMF] << 16);
      }
      acc[0][j] = __builtin_amdgcn_wmma_f32_16x16x32_bf16(
                      false, a[0].v, false, b.v, (short)0, acc[0][j], false, false);
      acc[1][j] = __builtin_amdgcn_wmma_f32_16x16x32_bf16(
                      false, a[1].v, false, b.v, (short)0, acc[1][j], false, false);
    }
  }
  __syncthreads();   // everyone done reading sA before it becomes sC

  // epilogue in two 16-row phases: scatter C tiles to LDS, then
  // out = y @ blockdiag(D)^T (forward rotation), f32 to HBM.
  const int nn = lane & 15;
  const int m8 = (lane >> 4) << 3;
  for (int mt = 0; mt < 2; ++mt) {
    // scatter (ISA C layout: N=lane&15, M=v + 8*(lane>=16))
#pragma unroll
    for (int j = 0; j < 4; ++j) {
      int n0 = ((wave << 2) + j) << 4;
#pragma unroll
      for (int v = 0; v < 8; ++v)
        sC[(m8 + v) * LDCH + n0 + nn] = acc[mt][j][v];
    }
    __syncthreads();

    for (int c = tid; c < 16 * 128; c += NTHR) {
      int r16 = c >> 7, ch = c & 127;
      int r = mt * 16 + r16;              // row within block
      int l = ch >> 5, u = ch & 31;
      int dd = 2 * l + 1;
      int base = lbase[l] + u * dd;
      const float* yr = sC + r16 * LDCH + base;
      float* orow = out + (size_t)(row0 + r) * DIMF + base;
      if (l == 0) {
        orow[0] = yr[0];
      } else {
        const float* Db = sD + r * 84 + doff[l];
        float yin[7];
        for (int j = 0; j < dd; ++j) yin[j] = yr[j];
        for (int i2 = 0; i2 < dd; ++i2) {
          float s = 0.0f;
          for (int j = 0; j < dd; ++j) s += Db[i2 * dd + j] * yin[j];
          orow[i2] = s;
        }
      }
    }
    __syncthreads();   // phase buffer reuse
  }
}

// ---------------- host launcher ----------------
extern "C" void kernel_launch(void* const* d_in, const int* in_sizes, int n_in,
                              void* d_out, int out_size, void* d_ws, size_t ws_size,
                              hipStream_t stream) {
  const float* x   = (const float*)d_in[0];   // [N,512]
  const float* vec = (const float*)d_in[1];   // [N,3]
  const float* W   = (const float*)d_in[2];   // [512,512]
  float* out       = (float*)d_out;
  const int n = in_sizes[0] / DIMF;

  // ws layout: [0, 512KB): W bf16 ; [512KB, +N*84*4): D blocks
  unsigned short* Wb = (unsigned short*)d_ws;
  float* Dws = (float*)((char*)d_ws + (size_t)DIMF * DIMF * 2);

  wcvt_kernel<<<(DIMF * DIMF) / NTHR, NTHR, 0, stream>>>(W, Wb);
  wigner_kernel<<<(n + NTHR - 1) / NTHR, NTHR, 0, stream>>>(vec, Dws, n);
  fused_kernel<<<(n + ROWS - 1) / ROWS, NTHR, 0, stream>>>(x, Wb, Dws, out, n);
}
